// Attention_Layer_66589172957625
// MI455X (gfx1250) — matrix operand order repaired and checked by reading, not compiled, not run on hardware
//
#include <hip/hip_runtime.h>
#include <hip/hip_bf16.h>

// Problem constants (match reference: B=8, N=4096, D=128)
#define BB 8
#define NN 4096
#define DD 128
#define QBLK 128   // query rows per workgroup (8 waves x 16 rows)
#define KBLK 64    // key block per staged LDS iteration

typedef __attribute__((ext_vector_type(16))) __bf16        v16bf;
typedef __attribute__((ext_vector_type(8)))  float         v8f;
typedef __attribute__((ext_vector_type(4)))  unsigned int  u32x4;
typedef __attribute__((ext_vector_type(4)))  int           i32x4;

union V16BF {
  v16bf v;
  u32x4 q[2];
};

static __device__ __forceinline__ v8f v8f_zero() {
  v8f r;
#pragma unroll
  for (int i = 0; i < 8; ++i) r[i] = 0.0f;
  return r;
}

static __device__ __forceinline__ v8f wmma_bf16(v16bf a, v16bf b, v8f c) {
  // D = A(16x32 bf16) x B(32x16 bf16) + C(16x16 f32)
  return __builtin_amdgcn_wmma_f32_16x16x32_bf16(
      /*neg_a=*/false, a, /*neg_b=*/false, b,
      /*c_mod=*/(short)0, c, /*reuse_a=*/false, /*reuse_b=*/false);
}

// ---- 16-byte global -> LDS copy: async DMA path (confirmed present).
#if defined(__has_builtin)
#if __has_builtin(__builtin_amdgcn_global_load_async_to_lds_b128)
#define HAS_ASYNC_LDS 1
#endif
#endif

static __device__ __forceinline__ void copy16_g2l(__bf16* dst_lds,
                                                  const __bf16* src_glob) {
#ifdef HAS_ASYNC_LDS
  __builtin_amdgcn_global_load_async_to_lds_b128(
      (__attribute__((address_space(1))) i32x4*)src_glob,
      (__attribute__((address_space(3))) i32x4*)dst_lds,
      /*offset=*/0, /*cpol=*/0);
#else
  *(u32x4*)dst_lds = *(const u32x4*)src_glob;
#endif
}

static __device__ __forceinline__ void wait_async_copies() {
#ifdef HAS_ASYNC_LDS
#if __has_builtin(__builtin_amdgcn_s_wait_asynccnt)
  __builtin_amdgcn_s_wait_asynccnt(0);
#else
  asm volatile("s_wait_asynccnt 0" ::: "memory");
#endif
#endif
}

// -----------------------------------------------------------------------------
// Kernel 1: Q/K/V projection.  y = x @ W^T, outputs stored as bf16.
// Q, K row-major [B][N][D]; V stored TRANSPOSED as Vt[B][D][N] so the attention
// kernel's V B-matrix tiles are contiguous loads.
// -----------------------------------------------------------------------------
__global__ __launch_bounds__(256) void qkv_proj_kernel(
    const float* __restrict__ X,  const float* __restrict__ Wq,
    const float* __restrict__ Wk, const float* __restrict__ Wv,
    __bf16* __restrict__ Qo, __bf16* __restrict__ Ko, __bf16* __restrict__ Vto)
{
  const int tile = blockIdx.x;                 // B*N/16 tiles
  const int b    = tile / (NN / 16);
  const int row0 = (tile % (NN / 16)) * 16;
  const int lane = threadIdx.x & 31;
  const int wave = threadIdx.x >> 5;
  const int ln   = lane & 15;
  const int hi   = lane >> 4;

  // A tiles: 16 rows of X, 4 chunks of Kdim=32, native f32->bf16 converts.
  const float* xrow = X + ((size_t)b * NN + row0 + ln) * DD;
  V16BF xa[4];
#pragma unroll
  for (int c = 0; c < 4; ++c) {
    const int k0 = 32 * c + 8 * hi;
#pragma unroll
    for (int i = 0; i < 8; ++i) {
      xa[c].v[i]     = (__bf16)xrow[k0 + i];
      xa[c].v[i + 8] = (__bf16)xrow[k0 + 16 + i];
    }
  }

  const float* Wmats[3] = { Wq, Wk, Wv };
#pragma unroll
  for (int t = 0; t < 3; ++t) {
    const int e0 = 16 * wave + ln;             // feature column this lane covers
    const float* wrow = Wmats[t] + (size_t)e0 * DD;
    v8f c = v8f_zero();
#pragma unroll
    for (int cc = 0; cc < 4; ++cc) {
      const int d0 = 32 * cc + 8 * hi;         // B layout: lane = column n
      V16BF wb;
#pragma unroll
      for (int i = 0; i < 16; ++i) wb.v[i] = (__bf16)wrow[d0 + i + (i >= 8 ? 8 : 0)];
      c = wmma_bf16(xa[cc].v, wb.v, c);
    }
    // C layout: element (m = v + 8*hi, n = ln)
#pragma unroll
    for (int v = 0; v < 8; ++v) {
      const int r = row0 + v + 8 * hi;
      const __bf16 val = (__bf16)c[v];
      if (t == 0)      Qo [((size_t)b * NN + r) * DD + e0] = val;
      else if (t == 1) Ko [((size_t)b * NN + r) * DD + e0] = val;
      else             Vto[((size_t)b * DD + e0) * NN + r] = val;  // transposed
    }
  }
}

// -----------------------------------------------------------------------------
// Kernel 2: flash attention (unscaled scores, softmax over keys).
// 256 workgroups = B x (N/128); 8 waves x 16 query rows each.
// 64-key blocks staged into LDS via async DMA, double buffered: one barrier
// per 32 WMMAs.  Per block: 16 WMMA for S = Q K^T (4 tiles), online softmax,
// LDS bounce to A-layout bf16 P, 16 WMMA for O += P V.
// -----------------------------------------------------------------------------
__global__ __launch_bounds__(256) void flash_attn_kernel(
    const __bf16* __restrict__ Q,  const __bf16* __restrict__ K,
    const __bf16* __restrict__ Vt, float* __restrict__ Out)
{
  const int wg    = blockIdx.x;              // B * (N/QBLK)
  const int b     = wg / (NN / QBLK);
  const int q0    = (wg % (NN / QBLK)) * QBLK;
  const int tid   = threadIdx.x;
  const int lane  = tid & 31;
  const int wave  = tid >> 5;
  const int ln    = lane & 15;
  const int hi    = lane >> 4;
  const int qrow0 = q0 + wave * 16;

  __shared__ __bf16 Kbuf[2][KBLK][DD];        // 32 KB  (keys x d)
  __shared__ __bf16 Vbuf[2][DD][KBLK];        // 32 KB  (d x keys)
  __shared__ __bf16 pstage[8][16][KBLK];      // 16 KB  per-wave P tiles

  const __bf16* Qb = Q  + ((size_t)b * NN + qrow0) * DD;
  const __bf16* Kb = K  + (size_t)b * NN * DD;
  const __bf16* Vb = Vt + (size_t)b * DD * NN;   // [D][N]

  // Cooperative staging of one 64-key block (K: 16KB, V: 16KB => 128B/thread,
  // 8 async b128 DMAs per thread).
  auto issue_stage = [&](int buf, int kb) {
    {                                          // K block: row-major [kk][d]
      const int kk = tid >> 2, d0 = (tid & 3) << 5;   // 32 bf16 = 64B run
      const __bf16* src = Kb + (size_t)(kb + kk) * DD + d0;
#pragma unroll
      for (int o = 0; o < 32; o += 8)
        copy16_g2l(&Kbuf[buf][kk][d0 + o], src + o);
    }
    {                                          // V block: [d][kk] (transposed)
      const int d = tid >> 1, n0 = (tid & 1) << 5;
      const __bf16* src = Vb + (size_t)d * NN + kb + n0;
#pragma unroll
      for (int o = 0; o < 32; o += 8)
        copy16_g2l(&Vbuf[buf][d][n0 + o], src + o);
    }
  };

  // ---- Q A-tiles held in registers across the whole key loop (4 x v16bf).
  V16BF qa[4];
  {
    const __bf16* qrow = Qb + (size_t)ln * DD;
#pragma unroll
    for (int c = 0; c < 4; ++c) {
      const int k0 = 32 * c + 8 * hi;
      qa[c].q[0] = *(const u32x4*)(qrow + k0);
      qa[c].q[1] = *(const u32x4*)(qrow + k0 + 16);
    }
  }

  float rmax[8], rsum[8];
  v8f acc[8];
#pragma unroll
  for (int v = 0; v < 8; ++v) { rmax[v] = -3.0e38f; rsum[v] = 0.0f; }
#pragma unroll
  for (int j = 0; j < 8; ++j) acc[j] = v8f_zero();

  issue_stage(0, 0);                           // prologue prefetch

  for (int kb = 0; kb < NN; kb += KBLK) {
    const int cur = (kb >> 6) & 1;
    wait_async_copies();                       // my block-portion landed
    __syncthreads();                           // everyone's portion landed,
                                               // and prev block fully consumed
    if (kb + KBLK < NN) issue_stage(cur ^ 1, kb + KBLK);  // overlap next DMA

    // ---- S tiles: 4 x 16 keys, from LDS K block
    v8f s[4];
#pragma unroll
    for (int q4 = 0; q4 < 4; ++q4) s[q4] = v8f_zero();
#pragma unroll
    for (int c = 0; c < 4; ++c) {
      const int d0 = 32 * c + 8 * hi;          // B layout: lane=col, Kdim run
#pragma unroll
      for (int q4 = 0; q4 < 4; ++q4) {
        V16BF kt;
        kt.q[0] = *(const u32x4*)&Kbuf[cur][16 * q4 + ln][d0];
        kt.q[1] = *(const u32x4*)&Kbuf[cur][16 * q4 + ln][d0 + 16];
        s[q4] = wmma_bf16(qa[c].v, kt.v, s[q4]);
      }
    }

    // ---- online softmax per output row (C layout: row = v + 8*hi, col = ln)
#pragma unroll
    for (int v = 0; v < 8; ++v) {
      float t = fmaxf(fmaxf(s[0][v], s[1][v]), fmaxf(s[2][v], s[3][v]));
#pragma unroll
      for (int m = 1; m < 16; m <<= 1) t = fmaxf(t, __shfl_xor(t, m, 32));
      const float nm    = fmaxf(rmax[v], t);
      const float scale = __expf(rmax[v] - nm);
      rmax[v] = nm;
      float e[4];
      float rs = 0.0f;
#pragma unroll
      for (int q4 = 0; q4 < 4; ++q4) { e[q4] = __expf(s[q4][v] - nm); rs += e[q4]; }
#pragma unroll
      for (int m = 1; m < 16; m <<= 1) rs += __shfl_xor(rs, m, 32);
      rsum[v] = rsum[v] * scale + rs;
#pragma unroll
      for (int j = 0; j < 8; ++j) acc[j][v] *= scale;
      const int r = v + 8 * hi;
#pragma unroll
      for (int q4 = 0; q4 < 4; ++q4)
        pstage[wave][r][16 * q4 + ln] = (__bf16)e[q4];
    }

    // ---- reload P in A layout: two 16x32 A-tiles (wave-private staging)
    V16BF pa0, pa1;
    pa0.q[0] = *(const u32x4*)&pstage[wave][ln][8 * hi];
    pa0.q[1] = *(const u32x4*)&pstage[wave][ln][16 + 8 * hi];
    pa1.q[0] = *(const u32x4*)&pstage[wave][ln][32 + 8 * hi];
    pa1.q[1] = *(const u32x4*)&pstage[wave][ln][48 + 8 * hi];

    // ---- O += P @ V over the 8 D-column tiles (V tiles from LDS, transposed)
#pragma unroll
    for (int j = 0; j < 8; ++j) {
      const __bf16* vr0 = &Vbuf[cur][16 * j + ln][8 * hi];         // keys 0..31
      const __bf16* vr1 = &Vbuf[cur][16 * j + ln][32 + 8 * hi];    // keys 32..63
      V16BF vb0, vb1;
      vb0.q[0] = *(const u32x4*)(vr0);
      vb0.q[1] = *(const u32x4*)(vr0 + 16);
      vb1.q[0] = *(const u32x4*)(vr1);
      vb1.q[1] = *(const u32x4*)(vr1 + 16);
      acc[j] = wmma_bf16(pa0.v, vb0.v, acc[j]);
      acc[j] = wmma_bf16(pa1.v, vb1.v, acc[j]);
    }
  }

  // ---- epilogue: out = O / rowsum, f32 (one divide per row, then scale)
  float inv[8];
#pragma unroll
  for (int v = 0; v < 8; ++v) inv[v] = 1.0f / rsum[v];
  float* orow = Out + ((size_t)b * NN + qrow0) * DD;
#pragma unroll
  for (int j = 0; j < 8; ++j) {
#pragma unroll
    for (int v = 0; v < 8; ++v) {
      const int r = v + 8 * hi;
      orow[(size_t)r * DD + 16 * j + ln] = acc[j][v] * inv[v];
    }
  }
}

// -----------------------------------------------------------------------------
extern "C" void kernel_launch(void* const* d_in, const int* in_sizes, int n_in,
                              void* d_out, int out_size, void* d_ws, size_t ws_size,
                              hipStream_t stream) {
  const float* X  = (const float*)d_in[0];   // [B,N,D]
  const float* Wq = (const float*)d_in[1];   // [D,D]
  const float* Wk = (const float*)d_in[2];   // [D,D]
  const float* Wv = (const float*)d_in[3];   // [D,D]
  float* Out = (float*)d_out;                // [B,N,D]

  // Workspace: Q, K (row-major bf16) + V transposed (bf16)  -> 24 MB total
  __bf16* Qw = (__bf16*)d_ws;
  __bf16* Kw = Qw + (size_t)BB * NN * DD;
  __bf16* Vw = Kw + (size_t)BB * NN * DD;

  qkv_proj_kernel<<<BB * (NN / 16), 256, 0, stream>>>(X, Wq, Wk, Wv, Qw, Kw, Vw);
  flash_attn_kernel<<<BB * (NN / QBLK), 256, 0, stream>>>(Qw, Kw, Vw, Out);
}
